// LowRankLinear_35192962023845
// MI455X (gfx1250) — compile-verified, matrix operand order checked by reading, not run on hardware
//
#include <hip/hip_runtime.h>

typedef __attribute__((ext_vector_type(16))) __bf16 v16bf;
typedef __attribute__((ext_vector_type(8)))  float  v8f;
typedef __attribute__((ext_vector_type(4)))  int    v4i;

#define IN_F  4096
#define OUT_F 4096
#define RANK  256
#define TOK   8192

#define AS1 __attribute__((address_space(1)))
#define AS3 __attribute__((address_space(3)))

#if __has_builtin(__builtin_amdgcn_global_load_async_to_lds_b128) && \
    __has_builtin(__builtin_amdgcn_s_wait_asynccnt)
#define HAVE_ASYNC_LDS 1
#endif

union BFrag {
    v16bf v;
    unsigned short h[16];
    uint4 q[2];
};

// fp32 -> bf16 bits, round-to-nearest-even (scalar fallback)
__device__ __forceinline__ unsigned short f2bf_bits(float f) {
    union { float f; unsigned u; } c; c.f = f;
    unsigned r = c.u + 0x7FFFu + ((c.u >> 16) & 1u);
    return (unsigned short)(r >> 16);
}

// pack two fp32 into one dword of 2x bf16 (a -> [15:0], b -> [31:16])
__device__ __forceinline__ unsigned pack_bf16x2(float a, float b) {
#if __has_builtin(__builtin_amdgcn_cvt_pk_bf16_f32)
    typedef __attribute__((ext_vector_type(2))) __bf16 v2bf;
    v2bf r = __builtin_amdgcn_cvt_pk_bf16_f32(a, b);
    return __builtin_bit_cast(unsigned, r);
#else
    return (unsigned)f2bf_bits(a) | ((unsigned)f2bf_bits(b) << 16);
#endif
}

// ---------------------------------------------------------------------------
// Stage 0: one-time fp32 -> bf16 conversion of a weight matrix (V or U).
// ---------------------------------------------------------------------------
__global__ __launch_bounds__(256) void lrl_cvt_bf16(
    const float* __restrict__ src, unsigned short* __restrict__ dst, int n4)
{
    int i = blockIdx.x * blockDim.x + threadIdx.x;
    if (i < n4) {
        float4 v = ((const float4*)src)[i];
        uint2 p;
        p.x = pack_bf16x2(v.x, v.y);
        p.y = pack_bf16x2(v.z, v.w);
        ((uint2*)dst)[i] = p;
    }
}

// ---------------------------------------------------------------------------
// Stage 1: T[8192x256](bf16) = x @ V^T.
// Block: 512 threads = 16 waves. Block tile: M=64 x N=256 (full rank).
// Wave w owns N-tile w (16 cols) and computes 4 M-subtiles, so each B
// fragment feeds 4 WMMAs. x is converted to bf16 while staging into
// double-buffered LDS (64x64 chunk, 8 KB per buffer).
// ---------------------------------------------------------------------------
__global__ __launch_bounds__(512) void lrl_stage1_xVt(
    const float* __restrict__ x, const unsigned short* __restrict__ Vb,
    unsigned short* __restrict__ T)
{
    __shared__ __align__(16) unsigned short sA[2][64 * 64];

    const int m0   = blockIdx.x * 64;
    const int tid  = threadIdx.x;
    const int lane = tid & 31;
    const int wave = tid >> 5;
    const int row  = lane & 15;
    const int koff = (lane >> 4) << 3;       // 0 or 8
    const int n    = wave * 16 + row;        // rank column

    const unsigned short* vrow = Vb + (size_t)n * IN_F;

    // staging: thread -> 8 consecutive x elements (one row, 2 float4 loads)
    const int sidx = tid * 8;                // half-index into 64x64 chunk
    const int sr   = sidx >> 6;              // 0..63
    const int sc   = sidx & 63;
    const float* xsrc = x + (size_t)(m0 + sr) * IN_F + sc;

    auto stage = [&](int buf, int k0) {
        float4 a0 = *(const float4*)(xsrc + k0);
        float4 a1 = *(const float4*)(xsrc + k0 + 4);
        uint4 d;
        d.x = pack_bf16x2(a0.x, a0.y);
        d.y = pack_bf16x2(a0.z, a0.w);
        d.z = pack_bf16x2(a1.x, a1.y);
        d.w = pack_bf16x2(a1.z, a1.w);
        *(uint4*)&sA[buf][sidx] = d;
    };

    stage(0, 0);
    __syncthreads();

    v8f acc[4] = {};
    for (int c = 0; c < IN_F / 64; ++c) {
        const int k0 = c * 64;

        if (c + 1 < IN_F / 64) {
            stage((c + 1) & 1, k0 + 64);
            if (c + 2 < IN_F / 64)
                __builtin_prefetch(xsrc + k0 + 128, 0, 3);
        }

        const unsigned short* sbuf = sA[c & 1];
#pragma unroll
        for (int ks = 0; ks < 64; ks += 32) {
            BFrag b;
            b.q[0] = *(const uint4*)&vrow[k0 + ks + koff];
            b.q[1] = *(const uint4*)&vrow[k0 + ks + 16 + koff];

            // hoist all 4 A-fragment LDS loads so the waits batch
            BFrag a[4];
#pragma unroll
            for (int msub = 0; msub < 4; ++msub) {
                const int rbase = (msub * 16 + row) * 64;
                a[msub].q[0] = *(const uint4*)&sbuf[rbase + ks + koff];
                a[msub].q[1] = *(const uint4*)&sbuf[rbase + ks + 16 + koff];
            }
#pragma unroll
            for (int msub = 0; msub < 4; ++msub)
                acc[msub] = __builtin_amdgcn_wmma_f32_16x16x32_bf16(
                    false, a[msub].v, false, b.v, (short)0, acc[msub], false, false);
        }
        __syncthreads();
    }

    // D layout: VGPR r -> M = r + (lane>>4)*8, N = lane&15
    const int hi = (lane >> 4) * 8;
#pragma unroll
    for (int msub = 0; msub < 4; ++msub)
#pragma unroll
        for (int r = 0; r < 8; ++r)
            T[(size_t)(m0 + msub * 16 + hi + r) * RANK + n] = f2bf_bits(acc[msub][r]);
}

// ---------------------------------------------------------------------------
// Stage 2: out = T(bf16) @ U^T + bias.
// Block: 512 threads = 16 waves. Block tile: M=64 x N=256.
// The 64-row T tile is a contiguous 32 KB block -> loaded once into LDS via
// the gfx1250 async global->LDS path (ASYNCcnt) when available.
// K=256 fully unrolled; each B fragment feeds 4 WMMAs (4 M-subtiles).
// ---------------------------------------------------------------------------
__global__ __launch_bounds__(512) void lrl_stage2_TUt(
    const unsigned short* __restrict__ T, const unsigned short* __restrict__ Ub,
    const float* __restrict__ bias, float* __restrict__ out)
{
    __shared__ __align__(16) unsigned short sT[64 * RANK];   // 32 KB

    const int m0   = blockIdx.x * 64;
    const int o0   = blockIdx.y * 256;
    const int tid  = threadIdx.x;
    const int lane = tid & 31;
    const int wave = tid >> 5;
    const int row  = lane & 15;
    const int koff = (lane >> 4) << 3;
    const int n    = o0 + wave * 16 + row;   // output column

    // ---- stage the contiguous T tile (64x256 bf16 = 2048 x b128) into LDS
    {
        const uint4* g = (const uint4*)(T + (size_t)m0 * RANK);
        uint4* l = (uint4*)sT;
#ifdef HAVE_ASYNC_LDS
#pragma unroll
        for (int i = 0; i < 4; ++i) {
            const int o = tid + i * 512;
            __builtin_amdgcn_global_load_async_to_lds_b128(
                (AS1 v4i*)(g + o), (AS3 v4i*)(l + o), 0, 0);
        }
        __builtin_amdgcn_s_wait_asynccnt(0);
#else
#pragma unroll
        for (int i = 0; i < 4; ++i) {
            const int o = tid + i * 512;
            l[o] = g[o];
        }
#endif
        __syncthreads();
    }

    const unsigned short* urow = Ub + (size_t)n * RANK;

    v8f acc[4] = {};
#pragma unroll
    for (int ks = 0; ks < RANK; ks += 32) {
        BFrag b;
        b.q[0] = *(const uint4*)&urow[ks + koff];
        b.q[1] = *(const uint4*)&urow[ks + 16 + koff];

        BFrag a[4];
#pragma unroll
        for (int msub = 0; msub < 4; ++msub) {
            const int rbase = (msub * 16 + row) * RANK;
            a[msub].q[0] = *(const uint4*)&sT[rbase + ks + koff];
            a[msub].q[1] = *(const uint4*)&sT[rbase + ks + 16 + koff];
        }
#pragma unroll
        for (int msub = 0; msub < 4; ++msub)
            acc[msub] = __builtin_amdgcn_wmma_f32_16x16x32_bf16(
                false, a[msub].v, false, b.v, (short)0, acc[msub], false, false);
    }

    const float bval = bias[n];
    const int hi = (lane >> 4) * 8;
#pragma unroll
    for (int msub = 0; msub < 4; ++msub)
#pragma unroll
        for (int r = 0; r < 8; ++r)
            out[(size_t)(m0 + msub * 16 + hi + r) * OUT_F + n] = acc[msub][r] + bval;
}

// ---------------------------------------------------------------------------
extern "C" void kernel_launch(void* const* d_in, const int* in_sizes, int n_in,
                              void* d_out, int out_size, void* d_ws, size_t ws_size,
                              hipStream_t stream) {
    (void)in_sizes; (void)n_in; (void)out_size; (void)ws_size;
    const float* x    = (const float*)d_in[0];
    const float* U    = (const float*)d_in[1];
    const float* V    = (const float*)d_in[2];
    const float* bias = (const float*)d_in[3];
    float* out        = (float*)d_out;

    // workspace layout: T (4 MB) | Vb (2 MB) | Ub (2 MB)
    unsigned short* T  = (unsigned short*)d_ws;
    unsigned short* Vb = T  + (size_t)TOK * RANK;
    unsigned short* Ub = Vb + (size_t)RANK * IN_F;

    const int n4 = (RANK * IN_F) / 4;
    lrl_cvt_bf16<<<dim3((n4 + 255) / 256), 256, 0, stream>>>(V, Vb, n4);
    lrl_cvt_bf16<<<dim3((n4 + 255) / 256), 256, 0, stream>>>(U, Ub, n4);

    lrl_stage1_xVt<<<dim3(TOK / 64), 512, 0, stream>>>(x, Vb, T);
    lrl_stage2_TUt<<<dim3(TOK / 64, OUT_F / 256), 512, 0, stream>>>(T, Ub, bias, out);
}
